// YOLO_GNN_51049981281358
// MI455X (gfx1250) — matrix operand order repaired and checked by reading, not compiled, not run on hardware
//
#include <hip/hip_runtime.h>
#include <hip/hip_bf16.h>
#include <stdint.h>

typedef __attribute__((ext_vector_type(8)))  float        v8f;
typedef __attribute__((ext_vector_type(8)))  unsigned int v8u;
typedef __attribute__((ext_vector_type(16))) __bf16       v16bf;

#define WMMA_BF16(C_, A_, B_) \
  __builtin_amdgcn_wmma_f32_16x16x32_bf16(false, (A_), false, (B_), (short)0, (C_), false, false)

__device__ __forceinline__ unsigned short f2bf(float f) {
  unsigned int u = __float_as_uint(f);
  u += 0x7FFFu + ((u >> 16) & 1u);          // round-to-nearest-even
  return (unsigned short)(u >> 16);
}
__device__ __forceinline__ float bf2f(unsigned short h) {
  return __uint_as_float(((unsigned int)h) << 16);
}
__device__ __forceinline__ unsigned int pack2(float lo, float hi) {
  return (unsigned int)f2bf(lo) | ((unsigned int)f2bf(hi) << 16);
}

// ---------------- constants ----------------
#define NB     64
#define IMG    224
#define C1     32
#define H1     112
#define C2     64
#define H2     56
#define FEAT   1024
#define HID    512
#define OUTD   256
#define NC     80
#define KSEL   2
#define PIX2   (H2*H2)          // 3136, divisible by 16

// =================== conv1: direct, f32 -> bf16 ===================
__global__ __launch_bounds__(256) void k_conv1(const float* __restrict__ x,
                                               const float* __restrict__ w,
                                               const float* __restrict__ bias,
                                               unsigned short* __restrict__ out) {
  int idx = blockIdx.x * blockDim.x + threadIdx.x;
  if (idx >= NB * C1 * H1 * H1) return;
  int xo = idx % H1; int t = idx / H1;
  int yo = t % H1;  t /= H1;
  int c  = t % C1;  int b = t / C1;
  float acc = bias[c];
  #pragma unroll
  for (int ci = 0; ci < 3; ci++)
    #pragma unroll
    for (int ky = 0; ky < 3; ky++)
      #pragma unroll
      for (int kx = 0; kx < 3; kx++) {
        int yi = 2 * yo + ky, xi = 2 * xo + kx;   // SAME pad_lo = 0
        if (yi < IMG && xi < IMG)
          acc += x[((b * 3 + ci) * IMG + yi) * IMG + xi] * w[((c * 3 + ci) * 3 + ky) * 3 + kx];
      }
  out[idx] = f2bf(fmaxf(acc, 0.f));
}

// =================== prep: conv2 weights->bf16 (OIHW is already [N][K]); zero GAP ===================
__global__ __launch_bounds__(256) void k_prep(const float* __restrict__ w2,
                                              unsigned short* __restrict__ Wt,
                                              float* __restrict__ pooled) {
  int tid = blockIdx.x * blockDim.x + threadIdx.x;
  if (tid < C2 * 288) Wt[tid] = f2bf(w2[tid]);
  int t2 = tid - C2 * 288;
  if (t2 >= 0 && t2 < NB * C2) pooled[t2] = 0.f;
}

// =================== conv2: implicit-GEMM WMMA + fused GAP ===================
__global__ __launch_bounds__(128) void k_conv2(const unsigned short* __restrict__ c1,
                                               const unsigned short* __restrict__ Wt,
                                               const float* __restrict__ bias,
                                               float* __restrict__ pooled) {
  __shared__ unsigned short As[16][288];      // 16 pixels x K=288 (bf16)
  int tile = blockIdx.x;
  int b  = tile / (PIX2 / 16);
  int p0 = (tile % (PIX2 / 16)) * 16;
  int tid = threadIdx.x;

  for (int idx = tid; idx < 16 * 288; idx += 128) {
    int m = idx / 288, k = idx % 288;
    int c = k / 9, r = k % 9, ky = r / 3, kx = r % 3;
    int p = p0 + m, y = p / H2, xq = p % H2;
    int yi = 2 * y + ky, xi = 2 * xq + kx;
    unsigned short v = 0;
    if (yi < H1 && xi < H1) v = c1[((b * C1 + c) * H1 + yi) * H1 + xi];
    As[m][k] = v;
  }
  __syncthreads();

  int wave = tid >> 5, lane = tid & 31;
  int hi = lane >> 4, col = lane & 15;
  int n = wave * 16 + col;                     // output channel
  v8f acc = {};
  for (int ks = 0; ks < 9; ks++) {
    v8u ai, bi;
    #pragma unroll
    for (int i = 0; i < 8; i++) {
      int ka = ks * 32 + hi * 16 + (i << 1);                          // A: K consecutive per half
      ai[i] = *(const unsigned int*)&As[col][ka];
      int kb = ks * 32 + hi * 16 + (i << 1);                          // B: K consecutive per half
      bi[i] = *(const unsigned int*)(Wt + n * 288 + kb);
    }
    acc = WMMA_BF16(acc, __builtin_bit_cast(v16bf, ai), __builtin_bit_cast(v16bf, bi));
  }
  float bn = bias[n], s = 0.f;
  #pragma unroll
  for (int j = 0; j < 8; j++) s += fmaxf(acc[j] + bn, 0.f);
  atomicAdd(&pooled[b * C2 + n], s * (1.0f / (float)PIX2));
}

// =================== feats = relu(pooled @ fcf_w + b) ===================
__global__ __launch_bounds__(256) void k_feats(const float* __restrict__ pooled,
                                               const float* __restrict__ w,
                                               const float* __restrict__ bias,
                                               float* __restrict__ feats) {
  int tid = blockIdx.x * blockDim.x + threadIdx.x;
  if (tid >= NB * FEAT) return;
  int b = tid >> 10, j = tid & (FEAT - 1);
  float acc = bias[j];
  for (int c = 0; c < C2; c++) acc += pooled[b * C2 + c] * w[c * FEAT + j];
  feats[tid] = fmaxf(acc, 0.f);
}

// =================== logits = feats @ fcc_w + b ===================
__global__ __launch_bounds__(256) void k_logits(const float* __restrict__ feats,
                                                const float* __restrict__ w,
                                                const float* __restrict__ bias,
                                                float* __restrict__ logits) {
  int tid = blockIdx.x * blockDim.x + threadIdx.x;
  if (tid >= NB * NC) return;
  int b = tid / NC, c = tid % NC;
  float acc = bias[c];
  for (int f = 0; f < FEAT; f++) acc += feats[b * FEAT + f] * w[f * NC + c];
  logits[tid] = acc;
}

// =================== per-sample graph: top2, d2, kNN, GCN-norm, agg (bf16, padded) ===================
__global__ __launch_bounds__(128) void k_graph(const float* __restrict__ feats,
                                               const float* __restrict__ logits,
                                               int* __restrict__ top_idx,
                                               float* __restrict__ An_ws,
                                               unsigned short* __restrict__ aggb) {
  __shared__ float Xg[5][FEAT];
  __shared__ float red[128];
  __shared__ float d2s[5][5];
  __shared__ float An[5][5];
  __shared__ float dinv5[5];
  int b = blockIdx.x, tid = threadIdx.x;
  const float* f = feats + b * FEAT;

  for (int idx = tid; idx < 5 * FEAT; idx += 128) {
    int t = idx >> 10, i = idx & (FEAT - 1);
    float v = (t == 0) ? f[i] : ((i < 256) ? f[(t - 1) * 256 + i] : 0.f);
    Xg[t][i] = v;
  }
  __syncthreads();

  for (int pr = 0; pr < 25; pr++) {
    int t = pr / 5, s = pr % 5;
    float loc = 0.f;
    for (int i = tid; i < FEAT; i += 128) { float d = Xg[t][i] - Xg[s][i]; loc += d * d; }
    red[tid] = loc; __syncthreads();
    for (int off = 64; off; off >>= 1) { if (tid < off) red[tid] += red[tid + off]; __syncthreads(); }
    if (tid == 0) d2s[t][s] = red[0];
    __syncthreads();
  }

  if (tid == 0) {   // top-2 logits, first-index tie break (matches lax.top_k)
    const float* lg = logits + b * NC;
    int i1 = 0; float v1 = lg[0];
    for (int i = 1; i < NC; i++) if (lg[i] > v1) { v1 = lg[i]; i1 = i; }
    int i2 = 0; float v2 = -3.4e38f;
    for (int i = 0; i < NC; i++) { if (i == i1) continue; if (lg[i] > v2) { v2 = lg[i]; i2 = i; } }
    top_idx[b * 2 + 0] = i1; top_idx[b * 2 + 1] = i2;
  }
  if (tid < 5) {    // kNN=4 (self included since d2 diag = 0), then A + I
    bool sel[5] = {false, false, false, false, false};
    for (int r = 0; r < 4; r++) {
      int bi = -1; float bd = 3.4e38f;
      for (int s = 0; s < 5; s++) { if (sel[s]) continue; if (d2s[tid][s] < bd) { bd = d2s[tid][s]; bi = s; } }
      sel[bi] = true;
    }
    for (int s = 0; s < 5; s++) An[tid][s] = (sel[s] ? 1.f : 0.f) + (s == tid ? 1.f : 0.f);
  }
  __syncthreads();
  if (tid < 5) {
    float dsum = 0.f;
    for (int s = 0; s < 5; s++) dsum += An[tid][s];
    dinv5[tid] = (dsum > 0.f) ? rsqrtf(dsum) : 0.f;
  }
  __syncthreads();
  if (tid < 25) {
    int t = tid / 5, s = tid % 5;
    float v = An[t][s] * dinv5[t] * dinv5[s];
    An[t][s] = v;
    An_ws[b * 25 + tid] = v;
  }
  __syncthreads();
  for (int idx = tid; idx < 16 * FEAT; idx += 128) {   // agg = An @ Xg, rows 5..15 zero-padded
    int t = idx >> 10, i = idx & (FEAT - 1);
    float acc = 0.f;
    if (t < 5) { for (int s = 0; s < 5; s++) acc += An[t][s] * Xg[s][i]; }
    aggb[(size_t)b * (16 * FEAT) + idx] = f2bf(acc);
  }
}

// =================== expert GNN: WMMA layer1 (K=1024) + propagate + WMMA layer2 (K=512) ===================
__global__ __launch_bounds__(256) void k_expert(const unsigned short* __restrict__ aggb,
                                                const int* __restrict__ top_idx,
                                                const float* __restrict__ w1,
                                                const float* __restrict__ b1,
                                                const float* __restrict__ w2,
                                                const float* __restrict__ b2,
                                                const float* __restrict__ An_ws,
                                                float* __restrict__ g) {
  __shared__ unsigned short h1s[16][HID];
  __shared__ unsigned short h2s[16][HID];
  __shared__ float Ans[25];
  int blk = blockIdx.x;
  int b = blk >> 1, kk = blk & 1;
  int tid = threadIdx.x;
  if (tid < 25) Ans[tid] = An_ws[b * 25 + tid];
  int e = top_idx[b * 2 + kk];
  int wave = tid >> 5, lane = tid & 31;
  int hi = lane >> 4, col = lane & 15;

  // -------- layer 1: h1 = relu(agg(16x1024) @ w1(1024x512) + b1) --------
  const unsigned short* arow = aggb + (size_t)b * (16 * FEAT) + (size_t)col * FEAT;
  const float* W1 = w1 + (size_t)e * FEAT * HID;
  v8f acc[4] = {};
  for (int ks = 0; ks < FEAT / 32; ks++) {
    if (ks + 1 < FEAT / 32)
      __builtin_prefetch(W1 + (size_t)((ks + 1) * 32 + lane) * HID, 0, 0);
    v8u ai;
    #pragma unroll
    for (int i = 0; i < 8; i++) {
      int ka = ks * 32 + hi * 16 + (i << 1);
      ai[i] = *(const unsigned int*)(arow + ka);
    }
    v16bf av = __builtin_bit_cast(v16bf, ai);
    #pragma unroll
    for (int t = 0; t < 4; t++) {
      int n = (wave + t * 8) * 16 + col;
      v8u bi;
      #pragma unroll
      for (int i = 0; i < 8; i++) {
        int kb = ks * 32 + hi * 16 + (i << 1);
        const float* p = W1 + (size_t)kb * HID + n;
        bi[i] = pack2(p[0], p[HID]);
      }
      acc[t] = WMMA_BF16(acc[t], av, __builtin_bit_cast(v16bf, bi));
    }
  }
  #pragma unroll
  for (int t = 0; t < 4; t++) {
    int n = (wave + t * 8) * 16 + col;
    float bb = b1[e * HID + n];
    #pragma unroll
    for (int j = 0; j < 8; j++) {
      int m = j + hi * 8;
      h1s[m][n] = f2bf(fmaxf(acc[t][j] + bb, 0.f));
    }
  }
  __syncthreads();

  // -------- propagate: h2 = An @ h1 (only rows 0..4 valid; pad rest with zero) --------
  for (int idx = tid; idx < 16 * HID; idx += 256) {
    int t = idx >> 9, n = idx & (HID - 1);
    float a = 0.f;
    if (t < 5) { for (int s = 0; s < 5; s++) a += Ans[t * 5 + s] * bf2f(h1s[s][n]); }
    h2s[t][n] = f2bf(a);
  }
  __syncthreads();

  // -------- layer 2: out = h2(16x512) @ w2(512x256) + b2; mean over 5 nodes --------
  const float* W2 = w2 + (size_t)e * HID * OUTD;
  v8f accB[2] = {};
  for (int ks = 0; ks < HID / 32; ks++) {
    v8u ai;
    #pragma unroll
    for (int i = 0; i < 8; i++) {
      int ka = ks * 32 + hi * 16 + (i << 1);
      ai[i] = *(const unsigned int*)&h2s[col][ka];
    }
    v16bf av = __builtin_bit_cast(v16bf, ai);
    #pragma unroll
    for (int t = 0; t < 2; t++) {
      int n = (wave + t * 8) * 16 + col;
      v8u bi;
      #pragma unroll
      for (int i = 0; i < 8; i++) {
        int kb = ks * 32 + hi * 16 + (i << 1);
        const float* p = W2 + (size_t)kb * OUTD + n;
        bi[i] = pack2(p[0], p[OUTD]);
      }
      accB[t] = WMMA_BF16(accB[t], av, __builtin_bit_cast(v16bf, bi));
    }
  }
  if (hi == 0) {   // rows 0..4 live in VGPRs j=0..4 of lanes 0-15
    #pragma unroll
    for (int t = 0; t < 2; t++) {
      int o = (wave + t * 8) * 16 + col;
      float s5 = accB[t][0] + accB[t][1] + accB[t][2] + accB[t][3] + accB[t][4];
      g[(b * 2 + kk) * OUTD + o] = s5 * 0.2f + b2[e * OUTD + o];
    }
  }
}

// =================== cw = comb(128x256) @ fin_w(256x80) ===================
__global__ __launch_bounds__(256) void k_cw(const float* __restrict__ g,
                                            const float* __restrict__ fin_w,
                                            float* __restrict__ cw) {
  int tid = blockIdx.x * blockDim.x + threadIdx.x;
  if (tid >= NB * KSEL * NC) return;
  int r = tid / NC, c = tid % NC;
  float acc = 0.f;
  for (int s = 0; s < OUTD; s++) acc += g[r * OUTD + s] * fin_w[s * NC + c];
  cw[tid] = acc;
}

// =================== final GCN (reuses sample 63's normalized 5x5 block) + pair mean ===================
__global__ __launch_bounds__(256) void k_final(const float* __restrict__ cw,
                                               const float* __restrict__ An_ws,
                                               const float* __restrict__ fin_b,
                                               float* __restrict__ out) {
  int tid = blockIdx.x * blockDim.x + threadIdx.x;
  if (tid >= NB * NC) return;
  int b = tid / NC, c = tid % NC;
  const float* An63 = An_ws + 63 * 25;
  float s = 0.f;
  for (int kk = 0; kk < 2; kk++) {
    int r = 2 * b + kk;
    float v;
    if (r < 5) {
      v = 0.f;
      for (int s2 = 0; s2 < 5; s2++) v += An63[r * 5 + s2] * cw[s2 * NC + c];
    } else {
      v = cw[r * NC + c];
    }
    s += v + fin_b[c];
  }
  out[tid] = 0.5f * s;
}

// =================== launch ===================
extern "C" void kernel_launch(void* const* d_in, const int* in_sizes, int n_in,
                              void* d_out, int out_size, void* d_ws, size_t ws_size,
                              hipStream_t stream) {
  const float* x    = (const float*)d_in[0];
  const float* c1w  = (const float*)d_in[1];
  const float* c1b  = (const float*)d_in[2];
  const float* c2w  = (const float*)d_in[3];
  const float* c2b  = (const float*)d_in[4];
  const float* fcfw = (const float*)d_in[5];
  const float* fcfb = (const float*)d_in[6];
  const float* fccw = (const float*)d_in[7];
  const float* fccb = (const float*)d_in[8];
  const float* gw1  = (const float*)d_in[9];
  const float* gb1  = (const float*)d_in[10];
  const float* gw2  = (const float*)d_in[11];
  const float* gb2  = (const float*)d_in[12];
  const float* finw = (const float*)d_in[13];
  const float* finb = (const float*)d_in[14];
  float* out = (float*)d_out;

  char* ws = (char*)d_ws;
  size_t off = 0;
  auto alloc = [&](size_t bytes) -> void* {
    void* p = ws + off;
    off += (bytes + 255) & ~(size_t)255;
    return p;
  };
  unsigned short* c1out  = (unsigned short*)alloc((size_t)NB * C1 * H1 * H1 * 2);
  unsigned short* Wt     = (unsigned short*)alloc((size_t)C2 * 288 * 2);
  float*          pooled = (float*)alloc((size_t)NB * C2 * 4);
  float*          feats  = (float*)alloc((size_t)NB * FEAT * 4);
  float*          logits = (float*)alloc((size_t)NB * NC * 4);
  int*            topi   = (int*)alloc((size_t)NB * 2 * 4);
  float*          Anw    = (float*)alloc((size_t)NB * 25 * 4);
  unsigned short* aggb   = (unsigned short*)alloc((size_t)NB * 16 * FEAT * 2);
  float*          gbuf   = (float*)alloc((size_t)NB * 2 * OUTD * 4);
  float*          cw     = (float*)alloc((size_t)NB * 2 * NC * 4);

  k_conv1 <<<(NB * C1 * H1 * H1 + 255) / 256, 256, 0, stream>>>(x, c1w, c1b, c1out);
  k_prep  <<<(C2 * 288 + NB * C2 + 255) / 256, 256, 0, stream>>>(c2w, Wt, pooled);
  k_conv2 <<<NB * (PIX2 / 16), 128, 0, stream>>>(c1out, Wt, c2b, pooled);
  k_feats <<<(NB * FEAT + 255) / 256, 256, 0, stream>>>(pooled, fcfw, fcfb, feats);
  k_logits<<<(NB * NC + 255) / 256, 256, 0, stream>>>(feats, fccw, fccb, logits);
  k_graph <<<NB, 128, 0, stream>>>(feats, logits, topi, Anw, aggb);
  k_expert<<<NB * KSEL, 256, 0, stream>>>(aggb, topi, gw1, gb1, gw2, gb2, Anw, gbuf);
  k_cw    <<<(NB * KSEL * NC + 255) / 256, 256, 0, stream>>>(gbuf, finw, cw);
  k_final <<<(NB * NC + 255) / 256, 256, 0, stream>>>(cw, Anw, finb, out);
}
